// SREL_intra_rep_rho_tester_30777735643694
// MI455X (gfx1250) — matrix operand order repaired and checked by reading, not compiled, not run on hardware
//
#include <hip/hip_runtime.h>
#include <hip/hip_bf16.h>
#include <math.h>

// ---------------- problem constants ----------------
#define B_SZ   64
#define LS     256
#define MM     16
#define NSTEP  10
#define HH     1024
#define DD     1280     // 2*Ls + 2*Lw + Ly

typedef __attribute__((ext_vector_type(16))) __bf16 v16bf;
typedef __attribute__((ext_vector_type(8)))  __bf16 bf16x8;
typedef __attribute__((ext_vector_type(8)))  float  v8f;

// workspace layout (bytes)
#define OFF_PHI  0u
#define OFF_W1F  (65536u)                       // phi: 64*256*4
#define OFF_V1F  (OFF_W1F + 26214400u)          // W1f: 10*1280*1024*2
#define OFF_W2F  (OFF_V1F + 2621440u)           // V1f: 1280*1024*2
#define OFF_H    (OFF_W2F + 5242880u)           // W2f: 10*1024*256*2
#define OFF_HR   (OFF_H   + 2097152u)           // h:   64*16*1024*2

// output layout (floats): s_stack [64][11][256] complex64, then rho [64][10][16] f32
#define S_OUT_FLOATS (B_SZ * (NSTEP + 1) * LS * 2)

// -------------------------------------------------------------------------
// Prep: convert fp32 weights -> bf16 in WMMA B-fragment-major layout.
// Fragment (lane, j) of a 32(K)x16(N) bf16 B tile:
//   col = lane & 15 ;  k = kb*32 + (j>>3)*16 + (lane>>4)*8 + (j&7)
// W1f : [step][ht=64][kb=40][lane=32][j=16]
// V1f : [ht=64][kb=40][lane][j]
// W2f : [step][lt=16][kb=32][lane][j]
// -------------------------------------------------------------------------
__global__ void srel_prep_weights(const float* __restrict__ W1,
                                  const float* __restrict__ V1,
                                  const float* __restrict__ W2,
                                  __bf16* __restrict__ w1f,
                                  __bf16* __restrict__ v1f,
                                  __bf16* __restrict__ w2f) {
    const int N1 = NSTEP * 64 * 40 * 512;   // 13,107,200
    const int N2 = 64 * 40 * 512;           //  1,310,720
    const int N3 = NSTEP * 16 * 32 * 512;   //  2,621,440
    const int NT = N1 + N2 + N3;
    for (int i = blockIdx.x * blockDim.x + threadIdx.x; i < NT;
         i += gridDim.x * blockDim.x) {
        if (i < N1) {
            int j = i & 15, lane = (i >> 4) & 31;
            int tmp = i >> 9;
            int kb = tmp % 40; tmp /= 40;
            int ht = tmp % 64; int st = tmp / 64;
            int col = ht * 16 + (lane & 15);
            int k   = kb * 32 + (j >> 3) * 16 + (lane >> 4) * 8 + (j & 7);
            w1f[i] = (__bf16)W1[((size_t)st * DD + k) * HH + col];
        } else if (i < N1 + N2) {
            int ii = i - N1;
            int j = ii & 15, lane = (ii >> 4) & 31;
            int tmp = ii >> 9;
            int kb = tmp % 40; int ht = tmp / 40;
            int col = ht * 16 + (lane & 15);
            int k   = kb * 32 + (j >> 3) * 16 + (lane >> 4) * 8 + (j & 7);
            v1f[ii] = (__bf16)V1[(size_t)k * HH + col];
        } else {
            int ii = i - N1 - N2;
            int j = ii & 15, lane = (ii >> 4) & 31;
            int tmp = ii >> 9;
            int kb = tmp % 32; tmp /= 32;
            int lt = tmp % 16; int st = tmp / 16;
            int col = lt * 16 + (lane & 15);
            int k   = kb * 32 + (j >> 3) * 16 + (lane >> 4) * 8 + (j & 7);
            w2f[ii] = (__bf16)W2[((size_t)st * HH + k) * LS + col];
        }
    }
}

// -------------------------------------------------------------------------
// Kernel A: per step, grid (B=64, H/64=16), 256 threads = 8 wave32.
// Build x tile [16 x 1280] bf16 in LDS, then each wave computes one
// 16x16 output tile of h (waves 0-3, W1 path) or hr (waves 4-7, V1 path)
// with 40 v_wmma_f32_16x16x32_bf16, bias + ReLU, store bf16.
// -------------------------------------------------------------------------
__global__ void __launch_bounds__(256)
srel_fused_mlp1(const float* __restrict__ phi,
                const float* __restrict__ wMr,   // [B][Lw][M]
                const float* __restrict__ wMi,
                const float* __restrict__ yM,    // [Ly][M]
                const __bf16* __restrict__ w1f,
                const __bf16* __restrict__ v1f,
                const float* __restrict__ b1,    // [NSTEP][H]
                const float* __restrict__ c1,    // [H]
                __bf16* __restrict__ hbuf,       // [B][16][H]
                __bf16* __restrict__ hrbuf,
                int step) {
    const int b = blockIdx.x;
    const int hblk = blockIdx.y;          // 0..15 -> 64 columns of H
    const int t = threadIdx.x;
    __shared__ __bf16 xt[16][DD + 8];     // pad 8 bf16 -> conflict-free rows

    // dynamic (trig) part: cols [0,512), identical for all 16 rows
    {
        float p = phi[b * LS + t];
        __bf16 cv = (__bf16)cosf(p);
        __bf16 sv = (__bf16)sinf(p);
        #pragma unroll
        for (int r = 0; r < 16; ++r) { xt[r][t] = cv; xt[r][LS + t] = sv; }
    }
    // fixed part: cols [512,1280): w_r | w_i | y
    for (int idx = t; idx < 16 * 768; idx += 256) {
        int r = idx / 768, c = idx - r * 768;
        float v;
        if (c < 256)       v = wMr[((size_t)b * 256 + c) * MM + r];
        else if (c < 512)  v = wMi[((size_t)b * 256 + (c - 256)) * MM + r];
        else               v = yM[(size_t)(c - 512) * MM + r];
        xt[r][512 + c] = (__bf16)v;
    }
    __syncthreads();

    const int lane = t & 31, w = t >> 5;
    const int hi = lane >> 4, cl = lane & 15;
    const bool isV = (w >= 4);
    const int ht = hblk * 4 + (w & 3);

    const __bf16* wf = isV
        ? (v1f + (size_t)(ht * 40) * 512 + lane * 16)
        : (w1f + (size_t)((step * 64 + ht) * 40) * 512 + lane * 16);
    const float bias = isV ? c1[ht * 16 + cl] : b1[step * HH + ht * 16 + cl];

    v8f acc;
    #pragma unroll
    for (int v = 0; v < 8; ++v) acc[v] = bias;

    for (int kb = 0; kb < 40; ++kb) {
        const __bf16* xr = &xt[cl][kb * 32 + 8 * hi];
        bf16x8 a0 = *(const bf16x8*)(xr);
        bf16x8 a1 = *(const bf16x8*)(xr + 16);
        bf16x8 b0 = *(const bf16x8*)(wf);
        bf16x8 b1v = *(const bf16x8*)(wf + 8);
        v16bf A, Bf;
        #pragma unroll
        for (int j = 0; j < 8; ++j) {
            A[j] = a0[j]; A[8 + j] = a1[j];
            Bf[j] = b0[j]; Bf[8 + j] = b1v[j];
        }
        acc = __builtin_amdgcn_wmma_f32_16x16x32_bf16(
            false, A, false, Bf, (short)0, acc, false, false);
        wf += 512;
    }

    __bf16* outp = (isV ? hrbuf : hbuf) + (size_t)b * MM * HH + ht * 16 + cl;
    #pragma unroll
    for (int v = 0; v < 8; ++v) {
        float val = acc[v];
        val = val > 0.f ? val : 0.f;
        outp[(size_t)(v + 8 * hi) * HH] = (__bf16)val;
    }
}

// -------------------------------------------------------------------------
// Kernel B: per step, grid B=64, 256 threads = 8 waves.
// rho = sigmoid(hr·V2 + c2); eta = h@W2 + b2 via WMMA (2 tiles/wave);
// eta_net = sum_m rho[m]*eta[m][:]; phi -= eta_net; emit s(step), rho(step).
// -------------------------------------------------------------------------
__global__ void __launch_bounds__(256)
srel_gate_update(float* __restrict__ phi,
                 const __bf16* __restrict__ hbuf,
                 const __bf16* __restrict__ hrbuf,
                 const __bf16* __restrict__ w2f,
                 const float* __restrict__ b2,   // [NSTEP][Ls]
                 const float* __restrict__ V2,   // [H]
                 const float* __restrict__ c2,   // [1]
                 float* __restrict__ out,
                 int step) {
    const int b = blockIdx.x;
    const int t = threadIdx.x;
    const int lane = t & 31, w = t >> 5;
    __shared__ float phi_s[LS];
    __shared__ float part[256];
    __shared__ float rho_s[MM];
    __shared__ float etan[LS];

    // phi snapshot + s output for this step
    {
        float p = phi[b * LS + t];
        phi_s[t] = p;
        int o = ((b * (NSTEP + 1) + step) * LS + t) * 2;
        out[o]     = cosf(p);
        out[o + 1] = sinf(p);
    }
    // rho partial dot: thread (m = t>>4, p = t&15) covers 64 H elements
    {
        int m = t >> 4, p = t & 15;
        const __bf16* hr = hrbuf + ((size_t)b * MM + m) * HH + p * 64;
        float s = 0.f;
        #pragma unroll 8
        for (int i = 0; i < 64; ++i) s += (float)hr[i] * V2[p * 64 + i];
        part[t] = s;
    }
    __syncthreads();
    if (t < MM) {
        float s = 0.f;
        #pragma unroll
        for (int i = 0; i < 16; ++i) s += part[t * 16 + i];
        s += c2[0];
        float r = 1.f / (1.f + expf(-s));
        rho_s[t] = r;
        out[S_OUT_FLOATS + (b * NSTEP + step) * MM + t] = r;
    }
    __syncthreads();

    // eta tiles via WMMA; fold rho in-register, reduce across lane halves
    const int hi = lane >> 4, cl = lane & 15;
    const __bf16* hA = hbuf + (size_t)b * MM * HH + (size_t)cl * HH;
    #pragma unroll
    for (int i = 0; i < 2; ++i) {
        const int lt = w * 2 + i;
        const float bias = b2[step * LS + lt * 16 + cl];
        v8f acc;
        #pragma unroll
        for (int v = 0; v < 8; ++v) acc[v] = bias;
        const __bf16* w2p = w2f + (size_t)((step * 16 + lt) * 32) * 512 + lane * 16;
        for (int kb = 0; kb < 32; ++kb) {
            const __bf16* ar = hA + kb * 32 + 8 * hi;
            bf16x8 a0 = *(const bf16x8*)(ar);
            bf16x8 a1 = *(const bf16x8*)(ar + 16);
            bf16x8 b0 = *(const bf16x8*)(w2p);
            bf16x8 b1v = *(const bf16x8*)(w2p + 8);
            v16bf A, Bf;
            #pragma unroll
            for (int j = 0; j < 8; ++j) {
                A[j] = a0[j]; A[8 + j] = a1[j];
                Bf[j] = b0[j]; Bf[8 + j] = b1v[j];
            }
            acc = __builtin_amdgcn_wmma_f32_16x16x32_bf16(
                false, A, false, Bf, (short)0, acc, false, false);
            w2p += 512;
        }
        // c[v] = eta[v + 8*hi][lt*16 + cl]; gate by rho and sum rows
        float ps = 0.f;
        #pragma unroll
        for (int v = 0; v < 8; ++v) ps += rho_s[v + 8 * hi] * acc[v];
        ps += __shfl_xor(ps, 16, 32);   // wave32: combine row halves
        if (hi == 0) etan[lt * 16 + cl] = ps;
    }
    __syncthreads();
    phi[b * LS + t] = phi_s[t] - etan[t];
}

// final s from phi after 10 steps
__global__ void srel_final_s(const float* __restrict__ phi, float* __restrict__ out) {
    const int b = blockIdx.x, t = threadIdx.x;
    float p = phi[b * LS + t];
    int o = ((b * (NSTEP + 1) + NSTEP) * LS + t) * 2;
    out[o]     = cosf(p);
    out[o + 1] = sinf(p);
}

extern "C" void kernel_launch(void* const* d_in, const int* in_sizes, int n_in,
                              void* d_out, int out_size, void* d_ws, size_t ws_size,
                              hipStream_t stream) {
    (void)in_sizes; (void)n_in; (void)out_size; (void)ws_size;
    const float* phi_in = (const float*)d_in[0];
    const float* wMr    = (const float*)d_in[1];
    const float* wMi    = (const float*)d_in[2];
    const float* yM     = (const float*)d_in[3];
    const float* W1     = (const float*)d_in[4];
    const float* b1     = (const float*)d_in[5];
    const float* W2     = (const float*)d_in[6];
    const float* b2     = (const float*)d_in[7];
    const float* V1     = (const float*)d_in[8];
    const float* c1     = (const float*)d_in[9];
    const float* V2     = (const float*)d_in[10];
    const float* c2     = (const float*)d_in[11];
    float* out = (float*)d_out;
    char*  ws  = (char*)d_ws;

    float*  phi  = (float*)(ws + OFF_PHI);
    __bf16* w1f  = (__bf16*)(ws + OFF_W1F);
    __bf16* v1f  = (__bf16*)(ws + OFF_V1F);
    __bf16* w2f  = (__bf16*)(ws + OFF_W2F);
    __bf16* hbuf = (__bf16*)(ws + OFF_H);
    __bf16* hrbuf= (__bf16*)(ws + OFF_HR);

    hipMemcpyAsync(phi, phi_in, (size_t)B_SZ * LS * sizeof(float),
                   hipMemcpyDeviceToDevice, stream);
    srel_prep_weights<<<4096, 256, 0, stream>>>(W1, V1, W2, w1f, v1f, w2f);

    for (int s = 0; s < NSTEP; ++s) {
        srel_fused_mlp1<<<dim3(B_SZ, 16), 256, 0, stream>>>(
            phi, wMr, wMi, yM, w1f, v1f, b1, c1, hbuf, hrbuf, s);
        srel_gate_update<<<B_SZ, 256, 0, stream>>>(
            phi, hbuf, hrbuf, w2f, b2, V2, c2, out, s);
    }
    srel_final_s<<<B_SZ, 256, 0, stream>>>(phi, out);
}
